// EfficientAttention_5076651343922
// MI455X (gfx1250) — matrix operand order repaired and checked by reading, not compiled, not run on hardware
//
#include <hip/hip_runtime.h>

// ---------------------------------------------------------------------------
// Efficient (linear) attention for MI455X / gfx1250.
// f16 WMMA + f32 accum; Tensor Data Mover (TDM) staging with TENSORcnt
// pipelining. B=16, S=2305, E=768, H=12, D=64
// ---------------------------------------------------------------------------

#define BB 16
#define SS 2305
#define EE 768
#define HH 12
#define DD 64
#define MTOT (BB * SS)   // 36880
#define NKCH ((SS + 31) / 32)   // 73 K-chunks in state kernel

typedef __attribute__((ext_vector_type(16))) _Float16 v16h;
typedef __attribute__((ext_vector_type(8)))  _Float16 h8;
typedef __attribute__((ext_vector_type(8)))  float    v8f;
typedef __attribute__((ext_vector_type(4)))  unsigned int v4u;
typedef __attribute__((ext_vector_type(8)))  int      v8i;
typedef __attribute__((ext_vector_type(4)))  int      v4i;

union Frag16 { v16h v; h8 h[2]; _Float16 e[16]; };
union Acc8   { v8f  v; float    e[8];  };

__device__ __forceinline__ v8f wmma_f16(v16h a, v16h b, v8f c) {
  // 8 args: (neg_a, A, neg_b, B, c_mod, C, reuse_a, reuse_b)
  return __builtin_amdgcn_wmma_f32_16x16x32_f16(false, a, false, b,
                                                (short)0, c, false, false);
}

// ---------------------------------------------------------------------------
// TDM: 2-D tile load Global -> LDS (one per-wave DMA, tracked by TENSORcnt).
// Dims/strides in elements (data_size = 2 bytes). pad_ctl!=0 appends LDS
// padding: pad_interval bits[24:22], pad_amount bits[31:25], enable bit[20].
// D# packing per CDNA5 ISA.
// ---------------------------------------------------------------------------
__device__ __forceinline__ void tdm_load_2d_f16(
    unsigned lds_addr, const void* gaddr,
    unsigned tile_d0, unsigned tile_d1,
    unsigned tens_d0, unsigned tens_d1,
    unsigned stride0, unsigned pad_ctl) {
  unsigned long long ga = (unsigned long long)(size_t)gaddr;
  v4u g0;
  g0.x = 1u;                                   // count=1 (valid), user mode
  g0.y = lds_addr;                             // LDS byte address
  g0.z = (unsigned)ga;                         // global_addr[31:0]
  g0.w = (unsigned)(ga >> 32) | (2u << 30);    // global_addr[56:32] | type=2
  v8i g1;
  g1[0] = (int)((1u << 16) | pad_ctl);         // data_size=1 (2B) + pad ctl
  g1[1] = (int)((tens_d0 & 0xFFFFu) << 16);    // tensor_dim0[15:0] @bits 63:48
  g1[2] = (int)(((tens_d0 >> 16) & 0xFFFFu) | ((tens_d1 & 0xFFFFu) << 16));
  g1[3] = (int)(((tens_d1 >> 16) & 0xFFFFu) | ((tile_d0 & 0xFFFFu) << 16));
  g1[4] = (int)(tile_d1 & 0xFFFFu);            // tile_dim1 (tile_dim2 = 0)
  g1[5] = (int)stride0;                        // tensor_dim0_stride[31:0]
  g1[6] = 0;                                   // stride0[47:32]=0
  g1[7] = 0;
  v4i z4 = {0, 0, 0, 0};
#if defined(__clang_major__) && __clang_major__ >= 23
  v8i z8 = {0, 0, 0, 0, 0, 0, 0, 0};
  __builtin_amdgcn_tensor_load_to_lds(g0, g1, z4, z4, z8, 0);
#else
  __builtin_amdgcn_tensor_load_to_lds(g0, g1, z4, z4, 0);
#endif
}

// pad control for pitch-40-half rows (32 halves data + 8 halves pad):
// pad_enable(bit20) | pad_interval=3 (16 DWORDs = 64B) | pad_amount=3 (4 DWORDs)
#define PAD40 ((1u << 20) | (3u << 22) | (3u << 25))

// pi/2/S
__device__ const float ANGC = 1.57079632679489662f / 2305.0f;

// ---------------------------------------------------------------------------
// Kernel 0: convert the four fp32 weight matrices [768x768] to f16
// ---------------------------------------------------------------------------
__global__ __launch_bounds__(256) void pack_weights_kernel(
    const float* __restrict__ wq, const float* __restrict__ wk,
    const float* __restrict__ wv, const float* __restrict__ wo,
    _Float16* __restrict__ out) {
  const int NW = EE * EE;
  int idx = blockIdx.x * 256 + threadIdx.x;
  if (idx >= 4 * NW) return;
  int sel = idx / NW;
  int off = idx - sel * NW;
  const float* src = (sel == 0) ? wq : (sel == 1) ? wk : (sel == 2) ? wv : wo;
  out[idx] = (_Float16)src[off];
}

// ---------------------------------------------------------------------------
// Kernel 1: tiled GEMM  C[m,n] = sum_k X[m,k] * W[n,k]   (torch [out,in] W)
// Workgroup tile 128(M) x 128(N), BK=32, 256 threads = 8 waves.
// Weight tile staged by wave-0 TDM (double-buffered, pitch-40 pad), A tile
// staged manually (f32->f16 convert). DMA of chunk i+1 overlaps compute of i.
// MODE: 0=Q (f*trig*mask), 1=K (f*trig), 2=V (plain f16), 3=O (f32+bias)
// ---------------------------------------------------------------------------
template <int MODE, int AHALF>
__global__ __launch_bounds__(256) void proj_gemm(
    const float* __restrict__ Xf, const _Float16* __restrict__ Xh,
    const _Float16* __restrict__ W,
    _Float16* __restrict__ outA, _Float16* __restrict__ outB,
    float* __restrict__ outF, const float* __restrict__ bias,
    const int* __restrict__ mask) {
  __shared__ _Float16 As[128 * 40];      // pitch 40 halves
  __shared__ _Float16 Bs[2][128 * 40];   // double-buffered weight tile

  const int tid  = threadIdx.x;
  const int m0   = blockIdx.x * 128;
  const int n0   = blockIdx.y * 128;
  const int w    = tid >> 5;
  const int lane = tid & 31;
  const int hs   = lane >> 4;
  const int lm   = lane & 15;
  const int wm   = w >> 2;      // 0..1
  const int wn   = w & 3;       // 0..3

  v8f acc[4][2];
  for (int i = 0; i < 4; ++i)
    for (int j = 0; j < 2; ++j) acc[i][j] = v8f{};

  const int sr = tid >> 1;          // staging row 0..127
  const int sc = (tid & 1) * 16;    // staging col base {0,16}

  // prologue: DMA weight tile for k-chunk 0
  if (tid < 32) {
    tdm_load_2d_f16((unsigned)(size_t)&Bs[0][0], W + (size_t)n0 * EE,
                    32, 128, 32, 128, EE, PAD40);
  }

  for (int kidx = 0; kidx < EE / 32; ++kidx) {
    const int k0 = kidx * 32;
    // ---- stage A tile (128 x 32) manually ----
    {
      Frag16 t;
      const int gm = m0 + sr;
      if (gm < MTOT) {
        if (AHALF) {
          const h8* p = (const h8*)(Xh + (size_t)gm * EE + k0 + sc);
          t.h[0] = p[0]; t.h[1] = p[1];
        } else {
          const float* sp = Xf + (size_t)gm * EE + k0 + sc;
          float4 f0 = *(const float4*)(sp + 0);
          float4 f1 = *(const float4*)(sp + 4);
          float4 f2 = *(const float4*)(sp + 8);
          float4 f3 = *(const float4*)(sp + 12);
          t.e[0] = (_Float16)f0.x; t.e[1] = (_Float16)f0.y;
          t.e[2] = (_Float16)f0.z; t.e[3] = (_Float16)f0.w;
          t.e[4] = (_Float16)f1.x; t.e[5] = (_Float16)f1.y;
          t.e[6] = (_Float16)f1.z; t.e[7] = (_Float16)f1.w;
          t.e[8]  = (_Float16)f2.x; t.e[9]  = (_Float16)f2.y;
          t.e[10] = (_Float16)f2.z; t.e[11] = (_Float16)f2.w;
          t.e[12] = (_Float16)f3.x; t.e[13] = (_Float16)f3.y;
          t.e[14] = (_Float16)f3.z; t.e[15] = (_Float16)f3.w;
        }
      } else {
        for (int i = 0; i < 16; ++i) t.e[i] = (_Float16)0.0f;
      }
      *(h8*)(&As[sr * 40 + sc])     = t.h[0];
      *(h8*)(&As[sr * 40 + sc + 8]) = t.h[1];
    }
    // ---- wave 0: prefetch next weight tile, wait for current ----
    if (tid < 32) {
      if (kidx + 1 < EE / 32) {
        tdm_load_2d_f16((unsigned)(size_t)&Bs[(kidx + 1) & 1][0],
                        W + (size_t)n0 * EE + (k0 + 32),
                        32, 128, 32, 128, EE, PAD40);
        __builtin_amdgcn_s_wait_tensorcnt(1);   // current tile complete
      } else {
        __builtin_amdgcn_s_wait_tensorcnt(0);
      }
    }
    __syncthreads();

    const _Float16* Bcur = &Bs[kidx & 1][0];
    // ---- fragments ----
    v16h afr[4];
    for (int i = 0; i < 4; ++i) {
      const int row = wm * 64 + i * 16 + lm;
      Frag16 u;
      u.h[0] = *(const h8*)(&As[row * 40 + hs * 8]);        // K = hs*8 .. +7
      u.h[1] = *(const h8*)(&As[row * 40 + 16 + hs * 8]);   // K = 16+hs*8 ..
      afr[i] = u.v;
    }
    v16h bfr[2];
    for (int j = 0; j < 2; ++j) {
      const int col = wn * 32 + j * 16 + lm;
      Frag16 u;                                             // K = hs*8 .., 16+hs*8 ..
      u.h[0] = *(const h8*)(&Bcur[col * 40 + hs * 8]);
      u.h[1] = *(const h8*)(&Bcur[col * 40 + 16 + hs * 8]);
      bfr[j] = u.v;
    }
    for (int i = 0; i < 4; ++i)
      for (int j = 0; j < 2; ++j)
        acc[i][j] = wmma_f16(afr[i], bfr[j], acc[i][j]);
    __syncthreads();
  }

  // ---- epilogue ----
  for (int i = 0; i < 4; ++i) {
    for (int j = 0; j < 2; ++j) {
      Acc8 c; c.v = acc[i][j];
      const int n     = n0 + wn * 32 + j * 16 + lm;
      const int mbase = m0 + wm * 64 + i * 16 + hs * 8;
      for (int r = 0; r < 8; ++r) {
        const int m = mbase + r;
        if (m >= MTOT) continue;
        const float cv = c.e[r];
        const size_t o = (size_t)m * EE + n;
        if (MODE == 3) {
          outF[o] = cv + bias[n];
        } else if (MODE == 2) {
          outA[o] = (_Float16)cv;
        } else {
          const unsigned um = (unsigned)m;
          const unsigned bi = um / (unsigned)SS;
          const unsigned s  = um - bi * (unsigned)SS;
          float f = (cv < 0.0f) ? __expf(cv) : (cv + 1.0f);
          if (MODE == 0) f *= (mask[m] != 0) ? 1.0f : 0.0f;
          const float ang = ANGC * (float)s;
          outA[o] = (_Float16)(f * __cosf(ang));
          outB[o] = (_Float16)(f * __sinf(ang));
        }
      }
    }
  }
}

// ---------------------------------------------------------------------------
// Kernel 2: per-(b,h) state GEMMs:  state[kd,vd] = sum_s kX[s,kd] * vh[s,vd]
// One workgroup per bh; wave = {trig} x {16-row slice of kd}.
// All three 32x64 tiles per K-chunk fetched by TDM, double-buffered:
// chunk i+1's DMA overlaps chunk i's WMMAs. Tail rows zero-filled by TDM OOB
// (tensor_dim1 = S - s0).
// ---------------------------------------------------------------------------
__global__ __launch_bounds__(256) void state_kernel(
    const _Float16* __restrict__ kcos, const _Float16* __restrict__ ksin,
    const _Float16* __restrict__ vh, _Float16* __restrict__ stateOut) {
  __shared__ _Float16 kc[2][32 * 64];
  __shared__ _Float16 ks[2][32 * 64];
  __shared__ _Float16 vv[2][32 * 64];

  const int tid  = threadIdx.x;
  const int bh   = blockIdx.x;
  const int b    = bh / HH;
  const int h    = bh % HH;
  const int w    = tid >> 5;
  const int lane = tid & 31;
  const int hs   = lane >> 4;
  const int lm   = lane & 15;
  const int trig = w >> 2;            // 0 = cos, 1 = sin
  const int ms   = (w & 3) * 16;      // kd-row slice

  v8f acc[4];
  for (int j = 0; j < 4; ++j) acc[j] = v8f{};

  const size_t rowbase = (size_t)b * SS * EE + h * DD;

  // prologue: DMA chunk 0 into buffer 0
  if (tid < 32) {
    tdm_load_2d_f16((unsigned)(size_t)&kc[0][0], kcos + rowbase,
                    64, 32, 64, SS, EE, 0);
    tdm_load_2d_f16((unsigned)(size_t)&ks[0][0], ksin + rowbase,
                    64, 32, 64, SS, EE, 0);
    tdm_load_2d_f16((unsigned)(size_t)&vv[0][0], vh + rowbase,
                    64, 32, 64, SS, EE, 0);
  }

  for (int ci = 0; ci < NKCH; ++ci) {
    const int buf = ci & 1;
    if (tid < 32) {
      if (ci + 1 < NKCH) {
        const int sn = (ci + 1) * 32;
        const size_t gb = rowbase + (size_t)sn * EE;
        const unsigned rows = (unsigned)(SS - sn);   // OOB rows -> zeros
        tdm_load_2d_f16((unsigned)(size_t)&kc[buf ^ 1][0], kcos + gb,
                        64, 32, 64, rows, EE, 0);
        tdm_load_2d_f16((unsigned)(size_t)&ks[buf ^ 1][0], ksin + gb,
                        64, 32, 64, rows, EE, 0);
        tdm_load_2d_f16((unsigned)(size_t)&vv[buf ^ 1][0], vh + gb,
                        64, 32, 64, rows, EE, 0);
        __builtin_amdgcn_s_wait_tensorcnt(3);        // current 3 tiles done
      } else {
        __builtin_amdgcn_s_wait_tensorcnt(0);
      }
    }
    __syncthreads();

    const _Float16* asrc = trig ? &ks[buf][0] : &kc[buf][0];
    const _Float16* bsrc = &vv[buf][0];
    Frag16 ua;                                  // A[m=kd, k=s_local] = kX[s][kd]
    for (int i = 0; i < 16; ++i) {
      const int kk = ((i < 8) ? 0 : 16) + hs * 8 + (i & 7);
      ua.e[i] = asrc[kk * 64 + ms + lm];
    }
    for (int j = 0; j < 4; ++j) {
      Frag16 ub;                                // B[k=s_local, n=vd]
      for (int i = 0; i < 16; ++i) {
        const int kk = ((i < 8) ? 0 : 16) + hs * 8 + (i & 7);
        ub.e[i] = bsrc[kk * 64 + j * 16 + lm];
      }
      acc[j] = wmma_f16(ua.v, ub.v, acc[j]);
    }
    __syncthreads();
  }

  _Float16* out = stateOut + ((size_t)bh * 2 + trig) * (DD * DD);
  for (int j = 0; j < 4; ++j) {
    Acc8 c; c.v = acc[j];
    for (int r = 0; r < 8; ++r)
      out[(ms + r + 8 * hs) * DD + j * 16 + lm] = (_Float16)c.e[r];
  }
}

// ---------------------------------------------------------------------------
// Kernel 3: values[s,vd] = (q_cos[s,:] @ state_cos + q_sin[s,:] @ state_sin)/8
// grid.x = s-chunk of 128 rows, grid.y = bh. States staged by two TDM loads.
// ---------------------------------------------------------------------------
__global__ __launch_bounds__(256) void attn_kernel(
    const _Float16* __restrict__ qcos, const _Float16* __restrict__ qsin,
    const _Float16* __restrict__ states, _Float16* __restrict__ values) {
  __shared__ _Float16 sc[64 * 64];
  __shared__ _Float16 ss[64 * 64];

  const int tid   = threadIdx.x;
  const int chunk = blockIdx.x;
  const int bh    = blockIdx.y;
  const int b     = bh / HH;
  const int h     = bh % HH;
  const int w     = tid >> 5;
  const int lane  = tid & 31;
  const int hs    = lane >> 4;
  const int lm    = lane & 15;

  if (tid < 32) {
    tdm_load_2d_f16((unsigned)(size_t)&sc[0],
                    states + ((size_t)bh * 2 + 0) * (DD * DD),
                    64, 64, 64, 64, 64, 0);
    tdm_load_2d_f16((unsigned)(size_t)&ss[0],
                    states + ((size_t)bh * 2 + 1) * (DD * DD),
                    64, 64, 64, 64, 64, 0);
    __builtin_amdgcn_s_wait_tensorcnt(0);
  }
  __syncthreads();

  const int srow0 = chunk * 128 + w * 16;
  v8f acc[4];
  for (int j = 0; j < 4; ++j) acc[j] = v8f{};

  for (int trig = 0; trig < 2; ++trig) {
    const _Float16* qg = trig ? qsin : qcos;
    const _Float16* st = trig ? ss : sc;
    for (int kc2 = 0; kc2 < 2; ++kc2) {
      const int k0 = kc2 * 32;
      Frag16 ua;
      const int s = srow0 + lm;
      if (s < SS) {
        const _Float16* ap = qg + (size_t)(b * SS + s) * EE + h * DD + k0 + hs * 8;
        ua.h[0] = *(const h8*)ap;
        ua.h[1] = *(const h8*)(ap + 16);
      } else {
        for (int i = 0; i < 16; ++i) ua.e[i] = (_Float16)0.0f;
      }
      for (int j = 0; j < 4; ++j) {
        Frag16 ub;
        for (int i = 0; i < 16; ++i) {
          const int kk = k0 + ((i < 8) ? 0 : 16) + hs * 8 + (i & 7);
          ub.e[i] = st[kk * 64 + j * 16 + lm];
        }
        acc[j] = wmma_f16(ua.v, ub.v, acc[j]);
      }
    }
  }

  for (int j = 0; j < 4; ++j) {
    Acc8 c; c.v = acc[j];
    for (int r = 0; r < 8; ++r) {
      const int s = srow0 + r + 8 * hs;
      if (s < SS)
        values[(size_t)(b * SS + s) * EE + h * DD + j * 16 + lm] =
            (_Float16)(c.e[r] * 0.125f);   // /sqrt(64)
    }
  }
}

// ---------------------------------------------------------------------------
// Host side
// ---------------------------------------------------------------------------
extern "C" void kernel_launch(void* const* d_in, const int* in_sizes, int n_in,
                              void* d_out, int out_size, void* d_ws, size_t ws_size,
                              hipStream_t stream) {
  const float* q    = (const float*)d_in[0];
  const float* k    = (const float*)d_in[1];
  const float* v    = (const float*)d_in[2];
  const int*   mask = (const int*)d_in[3];
  const float* Wq   = (const float*)d_in[4];
  const float* Wk   = (const float*)d_in[5];
  const float* Wv   = (const float*)d_in[6];
  const float* Wo   = (const float*)d_in[7];
  const float* bo   = (const float*)d_in[8];
  float* out = (float*)d_out;

  // workspace layout (bytes)
  const size_t NW       = (size_t)EE * EE;                 // 589824 halves
  const size_t W_BYTES  = 4 * NW * sizeof(_Float16);
  const size_t ST_OFF   = W_BYTES;
  const size_t ST_BYTES = (size_t)BB * HH * 2 * DD * DD * sizeof(_Float16);
  const size_t ACT_OFF  = (size_t)8 << 20;                 // activations at 8 MiB
  const size_t ACT      = (size_t)MTOT * EE;               // halves per buffer
  const size_t NEEDED   = ACT_OFF + 5 * ACT * sizeof(_Float16);
  if (ws_size < NEEDED || ST_OFF + ST_BYTES > ACT_OFF) return;

  char* ws = (char*)d_ws;
  _Float16* wh     = (_Float16*)ws;
  _Float16* whq    = wh;
  _Float16* whk    = wh + NW;
  _Float16* whv    = wh + 2 * NW;
  _Float16* who    = wh + 3 * NW;
  _Float16* stateB = (_Float16*)(ws + ST_OFF);
  _Float16* qcosB  = (_Float16*)(ws + ACT_OFF);
  _Float16* qsinB  = qcosB + ACT;
  _Float16* kcosB  = qcosB + 2 * ACT;   // reused as `values` after state_kernel
  _Float16* ksinB  = qcosB + 3 * ACT;
  _Float16* vhB    = qcosB + 4 * ACT;
  _Float16* valB   = kcosB;

  pack_weights_kernel<<<(4 * (int)NW + 255) / 256, 256, 0, stream>>>(Wq, Wk, Wv, Wo, wh);

  dim3 g1((MTOT + 127) / 128, EE / 128);   // 289 x 6
  proj_gemm<0, 0><<<g1, 256, 0, stream>>>(q, nullptr, whq, qcosB, qsinB, nullptr, nullptr, mask);
  proj_gemm<1, 0><<<g1, 256, 0, stream>>>(k, nullptr, whk, kcosB, ksinB, nullptr, nullptr, nullptr);
  proj_gemm<2, 0><<<g1, 256, 0, stream>>>(v, nullptr, whv, vhB, nullptr, nullptr, nullptr, nullptr);

  state_kernel<<<BB * HH, 256, 0, stream>>>(kcosB, ksinB, vhB, stateB);

  attn_kernel<<<dim3((SS + 127) / 128, BB * HH), 256, 0, stream>>>(qcosB, qsinB, stateB, valB);

  proj_gemm<3, 1><<<g1, 256, 0, stream>>>(nullptr, valB, who, nullptr, nullptr, out, bo, nullptr);
}
